// RegionProposalNetwork_47665547051880
// MI455X (gfx1250) — matrix operand order repaired and checked by reading, not compiled, not run on hardware
//
#include <hip/hip_runtime.h>
#include <hip/hip_bf16.h>
#include <math.h>

// ---------------- types ----------------
typedef __attribute__((ext_vector_type(8)))  __bf16 v8bf;
typedef __attribute__((ext_vector_type(16))) __bf16 v16bf;
typedef __attribute__((ext_vector_type(8)))  float  v8f;
typedef __attribute__((ext_vector_type(4)))  unsigned int v4u;
typedef __attribute__((ext_vector_type(8)))  int v8i;
typedef __attribute__((ext_vector_type(4)))  int v4i;

#define RPN_A      9
#define RPN_PRE_N  6000
#define RPN_POST_N 300
#define RPN_P      36864      // 64*64*9 proposals per image
#define RPN_NPIX   16384      // 4*64*64 pixels total
#define RPN_K3     4608       // 9 taps * 512 cin (tap-major!)

#if defined(__AMDGCN__) && __has_builtin(__builtin_amdgcn_tensor_load_to_lds) && __has_builtin(__builtin_amdgcn_s_wait_tensorcnt)
#define USE_TDM 1
#else
#define USE_TDM 0
#endif

// ---------------- weight packing ----------------
// Wb[cout][tap*512 + cin] = w_conv[cout][cin][tap]  (tap = ky*3+kx)
__global__ void k_pack_w3(const float* __restrict__ w, __bf16* __restrict__ out) {
  int i = blockIdx.x * blockDim.x + threadIdx.x;
  if (i >= 512 * RPN_K3) return;
  int cout = i / RPN_K3;
  int rem  = i - cout * RPN_K3;
  int tap  = rem >> 9;
  int cin  = rem & 511;
  out[i] = (__bf16)w[((size_t)cout * 512 + cin) * 9 + tap];
}

__global__ void k_pack_small(const float* __restrict__ wcls, const float* __restrict__ wbbox,
                             const float* __restrict__ bcls, const float* __restrict__ bbbox,
                             __bf16* __restrict__ Wsm, float* __restrict__ bsm) {
  int i = blockIdx.x * blockDim.x + threadIdx.x;
  if (i < 64 * 512) {
    int m = i >> 9, k = i & 511;
    float v = 0.f;
    if (m < 18)      v = wcls[m * 512 + k];
    else if (m < 54) v = wbbox[(m - 18) * 512 + k];
    Wsm[i] = (__bf16)v;
  }
  if (i < 64) {
    float v = 0.f;
    if (i < 18)      v = bcls[i];
    else if (i < 54) v = bbbox[i - 18];
    bsm[i] = v;
  }
}

// ---------------- fragment helpers (ISA 7.12.2 layouts) ----------------
__device__ inline v16bf load_fragA(const __bf16* row, int kh) {
  v8bf lo = *(const v8bf*)(row + 8 * kh);
  v8bf hi = *(const v8bf*)(row + 16 + 8 * kh);
  return __builtin_shufflevector(lo, hi, 0,1,2,3,4,5,6,7,8,9,10,11,12,13,14,15);
}
__device__ inline v16bf load_fragB(const __bf16* row, int kh) {
  v8bf lo = *(const v8bf*)(row + 8 * kh);
  v8bf hi = *(const v8bf*)(row + 16 + 8 * kh);
  return __builtin_shufflevector(lo, hi, 0,1,2,3,4,5,6,7,8,9,10,11,12,13,14,15);
}

#define LDAB 40   // halfs per LDS row (80B, 16B aligned, conflict-skewed)

#if USE_TDM
// TDM: load 2D tile (32 halfs x 128 rows, row stride 4608 halfs) into LDS,
// padding 4 DWORDs after each 16 DWORDs -> LDS row stride = 40 halfs.
__device__ inline void tdm_load_a(unsigned lds_addr, const __bf16* gptr) {
  unsigned long long ga = (unsigned long long)(size_t)gptr;
  v4u g0;
  g0[0] = 1u;                                                  // count=1
  g0[1] = lds_addr;                                            // lds_addr
  g0[2] = (unsigned)ga;                                        // global_addr lo
  g0[3] = (unsigned)((ga >> 32) & 0x01FFFFFFu) | (2u << 30);   // ga hi | type=2
  v8i g1;
  g1[0] = (int)((1u << 16) | (1u << 20) | (3u << 22) | (3u << 25)); // 2B elems, pad 16dw->+4dw
  g1[1] = (int)(4608u << 16);    // tensor_dim0 = 4608
  g1[2] = (int)(512u << 16);     // tensor_dim1 = 512
  g1[3] = (int)(32u << 16);      // tile_dim0 = 32
  g1[4] = (int)128;              // tile_dim1 = 128, tile_dim2 = 0
  g1[5] = (int)4608;             // tensor_dim0_stride = 4608
  g1[6] = 0;
  g1[7] = 0;
  v4i z4 = {0, 0, 0, 0};
#if __clang_major__ >= 23
  v8i z8 = {0, 0, 0, 0, 0, 0, 0, 0};
  __builtin_amdgcn_tensor_load_to_lds(g0, g1, z4, z4, z8, 0);
#else
  __builtin_amdgcn_tensor_load_to_lds(g0, g1, z4, z4, 0);
#endif
}
#endif

// ---------------- 3x3 conv as implicit GEMM ----------------
// D[cout,n] = sum_k Wb[cout,k] * im2col(feat)[k,n], k = tap*512+cin (tap-major)
__global__ __launch_bounds__(256) void k_conv3(const float* __restrict__ feat,
                                               const __bf16* __restrict__ Wb,
                                               const float* __restrict__ bias,
                                               __bf16* __restrict__ mid) {
  __shared__ __bf16 lds_a[128 * LDAB];
  __shared__ __bf16 lds_b[128 * LDAB];
  const int tid  = threadIdx.x;
  const int lane = tid & 31;
  const int wv   = tid >> 5;
  const int wm   = wv & 1;
  const int wn   = wv >> 1;
  const int l15  = lane & 15;
  const int kh   = lane >> 4;
  const int bM   = blockIdx.y;
  const int bN   = blockIdx.x;

  v8f acc[4][2];
  #pragma unroll
  for (int i = 0; i < 4; ++i)
    #pragma unroll
    for (int j = 0; j < 2; ++j)
      #pragma unroll
      for (int e = 0; e < 8; ++e) acc[i][j][e] = 0.f;

  // per-thread invariants of the B gather
  const int kB   = tid >> 3;           // unused placeholder (kept simple below)
  (void)kB;

  for (int k0 = 0; k0 < RPN_K3; k0 += 32) {
    const int tap  = k0 >> 9;          // scalar: 0..8
    const int ky   = tap / 3;          // SALU
    const int kx   = tap - ky * 3;     // SALU
    const int cin0 = k0 & 511;

#if USE_TDM
    if (wv == 0)
      tdm_load_a((unsigned)(size_t)&lds_a[0],
                 Wb + (size_t)bM * 128 * RPN_K3 + k0);
#else
    {
      int c = tid;
      #pragma unroll
      for (int rep = 0; rep < 2; ++rep, c += 256) {
        int m = c >> 2, part = c & 3;
        *(v8bf*)&lds_a[m * LDAB + part * 8] =
            *(const v8bf*)(Wb + (size_t)(bM * 128 + m) * RPN_K3 + k0 + part * 8);
      }
    }
#endif
    // B tile: implicit im2col gather (tap fixed per K-step -> no divisions)
    #pragma unroll
    for (int rep = 0; rep < 16; ++rep) {
      int idx = tid + rep * 256;
      int k = idx >> 7, n = idx & 127;
      int cin = cin0 + k;
      int ng = bN * 128 + n;
      int b = ng >> 12, rm = ng & 4095;
      int y = rm >> 6, x = rm & 63;
      int iy = y + ky - 1, ix = x + kx - 1;
      float v = 0.f;
      if ((unsigned)iy < 64u && (unsigned)ix < 64u)
        v = feat[(((size_t)b * 512 + cin) * 64 + iy) * 64 + ix];
      lds_b[n * LDAB + k] = (__bf16)v;
    }
#if USE_TDM
    if (wv == 0) __builtin_amdgcn_s_wait_tensorcnt(0);
#endif
    __syncthreads();

    v16bf bf[2];
    #pragma unroll
    for (int nt = 0; nt < 2; ++nt)
      bf[nt] = load_fragB(&lds_b[(wn * 32 + nt * 16 + l15) * LDAB], kh);
    #pragma unroll
    for (int mt = 0; mt < 4; ++mt) {
      v16bf af = load_fragA(&lds_a[(wm * 64 + mt * 16 + l15) * LDAB], kh);
      #pragma unroll
      for (int nt = 0; nt < 2; ++nt)
        acc[mt][nt] = __builtin_amdgcn_wmma_f32_16x16x32_bf16(
            false, af, false, bf[nt], (short)0, acc[mt][nt], false, false);
    }
    __syncthreads();
  }

  #pragma unroll
  for (int mt = 0; mt < 4; ++mt)
    #pragma unroll
    for (int nt = 0; nt < 2; ++nt) {
      size_t ng = (size_t)bN * 128 + wn * 32 + nt * 16 + l15;
      #pragma unroll
      for (int r = 0; r < 8; ++r) {
        int cout = bM * 128 + wm * 64 + mt * 16 + r + 8 * kh;
        float v = acc[mt][nt][r] + bias[cout];
        v = v > 0.f ? v : 0.f;
        mid[(size_t)cout * RPN_NPIX + ng] = (__bf16)v;
      }
    }
}

// ---------------- fused 1x1 convs (cls 18 + bbox 36 -> M=64 padded) ----------------
__global__ __launch_bounds__(256) void k_conv1(const __bf16* __restrict__ mid,
                                               const __bf16* __restrict__ Wsm,
                                               const float* __restrict__ bsm,
                                               float* __restrict__ sc) {
  __shared__ __bf16 lds_a[64 * LDAB];
  __shared__ __bf16 lds_b[128 * LDAB];
  const int tid  = threadIdx.x;
  const int lane = tid & 31;
  const int wv   = tid >> 5;
  const int l15  = lane & 15;
  const int kh   = lane >> 4;
  const int bN   = blockIdx.x;

  v8f acc[4];
  #pragma unroll
  for (int i = 0; i < 4; ++i)
    #pragma unroll
    for (int e = 0; e < 8; ++e) acc[i][e] = 0.f;

  for (int k0 = 0; k0 < 512; k0 += 32) {
    {
      int m = tid >> 2, part = tid & 3;
      *(v8bf*)&lds_a[m * LDAB + part * 8] =
          *(const v8bf*)(Wsm + (size_t)m * 512 + k0 + part * 8);
    }
    #pragma unroll
    for (int rep = 0; rep < 16; ++rep) {
      int idx = tid + rep * 256;
      int k = idx >> 7, n = idx & 127;
      lds_b[n * LDAB + k] = mid[(size_t)(k0 + k) * RPN_NPIX + bN * 128 + n];
    }
    __syncthreads();
    v16bf bf = load_fragB(&lds_b[(wv * 16 + l15) * LDAB], kh);
    #pragma unroll
    for (int mt = 0; mt < 4; ++mt) {
      v16bf af = load_fragA(&lds_a[(mt * 16 + l15) * LDAB], kh);
      acc[mt] = __builtin_amdgcn_wmma_f32_16x16x32_bf16(
          false, af, false, bf, (short)0, acc[mt], false, false);
    }
    __syncthreads();
  }
  size_t n = (size_t)bN * 128 + wv * 16 + l15;
  #pragma unroll
  for (int mt = 0; mt < 4; ++mt)
    #pragma unroll
    for (int r = 0; r < 8; ++r) {
      int m = mt * 16 + r + 8 * kh;
      sc[(size_t)m * RPN_NPIX + n] = acc[mt][r] + bsm[m];
    }
}

// ---------------- softmax + anchor decode + clip + min-size ----------------
__global__ void k_props(const float* __restrict__ sc, const float* __restrict__ im_shape,
                        float* __restrict__ boxesAll, float* __restrict__ scoresAll) {
  int gid = blockIdx.x * blockDim.x + threadIdx.x;
  if (gid >= 4 * RPN_P) return;
  int b = gid / RPN_P;
  int p = gid - b * RPN_P;
  int nl = p / RPN_A, a = p - nl * RPN_A;
  int y = nl >> 6, x = nl & 63;
  int n = b * 4096 + nl;

  float sbg = sc[(size_t)a * RPN_NPIX + n];
  float sfg = sc[(size_t)(a + RPN_A) * RPN_NPIX + n];
  float prob = 1.f / (1.f + expf(sbg - sfg));

  float d0 = sc[(size_t)(18 + a * 4 + 0) * RPN_NPIX + n];
  float d1 = sc[(size_t)(18 + a * 4 + 1) * RPN_NPIX + n];
  float d2 = sc[(size_t)(18 + a * 4 + 2) * RPN_NPIX + n];
  float d3 = sc[(size_t)(18 + a * 4 + 3) * RPN_NPIX + n];

  int ridx = a / 3, sidx = a - ridx * 3;
  float ratio = (ridx == 0) ? 0.5f : (ridx == 1 ? 1.f : 2.f);
  float scale = (sidx == 0) ? 8.f : (sidx == 1 ? 16.f : 32.f);
  float wsr = rintf(sqrtf(256.f / ratio));
  float hsr = rintf(wsr * ratio);
  float aw = wsr * scale, ah = hsr * scale;
  float cx = (float)x * 16.f + 7.5f;
  float cy = (float)y * 16.f + 7.5f;

  float pcx = d0 * aw + cx;
  float pcy = d1 * ah + cy;
  float pw  = expf(d2) * aw;
  float ph  = expf(d3) * ah;
  float x1 = pcx - 0.5f * pw, y1 = pcy - 0.5f * ph;
  float x2 = pcx + 0.5f * pw, y2 = pcy + 0.5f * ph;

  float imh = im_shape[b * 2 + 0] - 1.f;
  float imw = im_shape[b * 2 + 1] - 1.f;
  x1 = fminf(fmaxf(x1, 0.f), imw);  x2 = fminf(fmaxf(x2, 0.f), imw);
  y1 = fminf(fmaxf(y1, 0.f), imh);  y2 = fminf(fmaxf(y2, 0.f), imh);
  float ww = x2 - x1 + 1.f, hh = y2 - y1 + 1.f;
  float s = (ww >= 16.f && hh >= 16.f) ? prob : -1e9f;

  float4 bb; bb.x = x1; bb.y = y1; bb.z = x2; bb.w = y2;
  *(float4*)(boxesAll + (size_t)gid * 4) = bb;
  scoresAll[gid] = s;
}

// ---------------- exact top-6000 via 4-pass radix select ----------------
__device__ inline unsigned fkey(float f) {
  unsigned u = __float_as_uint(f);
  return (u & 0x80000000u) ? ~u : (u | 0x80000000u);
}

__global__ __launch_bounds__(1024) void k_topk(const float* __restrict__ scoresAll,
                                               const float* __restrict__ boxesAll,
                                               float* __restrict__ topS,
                                               float* __restrict__ topB) {
  const int b = blockIdx.x;
  const float* s  = scoresAll + (size_t)b * RPN_P;
  const float* bx = boxesAll + (size_t)b * RPN_P * 4;
  __shared__ unsigned hist[256];
  __shared__ unsigned sh_prefix, sh_want;
  __shared__ unsigned cHi, cEq;
  const int tid = threadIdx.x;
  if (tid == 0) { sh_prefix = 0u; sh_want = RPN_PRE_N; cHi = 0u; cEq = 0u; }
  __syncthreads();

  for (int pass = 0; pass < 4; ++pass) {
    int shift = 24 - 8 * pass;
    if (tid < 256) hist[tid] = 0u;
    __syncthreads();
    unsigned prefix = sh_prefix;
    unsigned pmask  = pass ? (0xFFFFFFFFu << (32 - 8 * pass)) : 0u;
    for (int i = tid; i < RPN_P; i += 1024) {
      unsigned u = fkey(s[i]);
      if ((u & pmask) == (prefix & pmask))
        atomicAdd(&hist[(u >> shift) & 255u], 1u);
    }
    __syncthreads();
    if (tid == 0) {
      unsigned cum = 0, want = sh_want, sel = 0;
      for (int bin = 255; bin >= 0; --bin) {
        unsigned h = hist[bin];
        if (cum + h >= want) { sel = (unsigned)bin; sh_want = want - cum; break; }
        cum += h;
      }
      sh_prefix = prefix | (sel << shift);
    }
    __syncthreads();
  }
  const unsigned T = sh_prefix;
  const unsigned r = sh_want;
  const unsigned nHi = RPN_PRE_N - r;
  __syncthreads();

  for (int i = tid; i < RPN_P; i += 1024) {
    unsigned u = fkey(s[i]);
    int pos = -1;
    if (u > T)       pos = (int)atomicAdd(&cHi, 1u);
    else if (u == T) { unsigned e = atomicAdd(&cEq, 1u); if (e < r) pos = (int)(nHi + e); }
    if (pos >= 0) {
      topS[(size_t)b * RPN_PRE_N + pos] = s[i];
      *(float4*)(topB + ((size_t)b * RPN_PRE_N + pos) * 4) =
          *(const float4*)(bx + (size_t)i * 4);
    }
  }
}

// ---------------- greedy NMS, 300 steps, 1 WG per image ----------------
__global__ __launch_bounds__(1024) void k_nms(const float* __restrict__ topS,
                                              const float* __restrict__ topB,
                                              float* __restrict__ out) {
  const int b = blockIdx.x;
  __shared__ float sS[RPN_PRE_N];
  __shared__ float sA[RPN_PRE_N];
  __shared__ float redV[1024];
  __shared__ int   redI[1024];
  __shared__ float bb[4];
  __shared__ int   sbi, sbvalid;
  const float* B = topB + (size_t)b * RPN_PRE_N * 4;
  const int tid = threadIdx.x;

  for (int i = tid; i < RPN_PRE_N; i += 1024) {
    sS[i] = topS[(size_t)b * RPN_PRE_N + i];
    float x1 = B[i * 4], y1 = B[i * 4 + 1], x2 = B[i * 4 + 2], y2 = B[i * 4 + 3];
    sA[i] = (x2 - x1 + 1.f) * (y2 - y1 + 1.f);
  }
  __syncthreads();

  for (int t = 0; t < RPN_POST_N; ++t) {
    float mv = -1e30f; int mi = 0;
    for (int i = tid; i < RPN_PRE_N; i += 1024) {
      float v = sS[i];
      if (v > mv || (v == mv && i < mi)) { mv = v; mi = i; }
    }
    redV[tid] = mv; redI[tid] = mi;
    __syncthreads();
    for (int off = 512; off > 0; off >>= 1) {
      if (tid < off) {
        float v2 = redV[tid + off]; int i2 = redI[tid + off];
        if (v2 > redV[tid] || (v2 == redV[tid] && i2 < redI[tid])) {
          redV[tid] = v2; redI[tid] = i2;
        }
      }
      __syncthreads();
    }
    if (tid == 0) {
      int i = redI[0];
      sbi = i;
      sbvalid = (sS[i] > -1e8f) ? 1 : 0;
      bb[0] = B[i*4]; bb[1] = B[i*4+1]; bb[2] = B[i*4+2]; bb[3] = B[i*4+3];
    }
    __syncthreads();
    const int besti = sbi, valid = sbvalid;
    const float bx1 = bb[0], by1 = bb[1], bx2 = bb[2], by2 = bb[3];
    const float barea = sA[besti];
    if (valid) {
      for (int i = tid; i < RPN_PRE_N; i += 1024) {
        float xx1 = fmaxf(bx1, B[i*4]);
        float yy1 = fmaxf(by1, B[i*4+1]);
        float xx2 = fminf(bx2, B[i*4+2]);
        float yy2 = fminf(by2, B[i*4+3]);
        float iw = fmaxf(0.f, xx2 - xx1 + 1.f);
        float ih = fmaxf(0.f, yy2 - yy1 + 1.f);
        float inter = iw * ih;
        float iou = inter / (barea + sA[i] - inter);
        if (iou > 0.7f || i == besti) sS[i] = -1e9f;
      }
    }
    if (tid == 0) {
      float* o = out + ((size_t)b * RPN_POST_N + t) * 5;
      o[0] = (float)b;
      o[1] = valid ? bb[0] : 0.f;
      o[2] = valid ? bb[1] : 0.f;
      o[3] = valid ? bb[2] : 0.f;
      o[4] = valid ? bb[3] : 0.f;
    }
    __syncthreads();
  }
}

// ---------------- host launcher ----------------
extern "C" void kernel_launch(void* const* d_in, const int* in_sizes, int n_in,
                              void* d_out, int out_size, void* d_ws, size_t ws_size,
                              hipStream_t stream) {
  const float* base_feat = (const float*)d_in[0];
  const float* im_shape  = (const float*)d_in[1];
  const float* w_conv    = (const float*)d_in[3];
  const float* b_conv    = (const float*)d_in[4];
  const float* w_cls     = (const float*)d_in[5];
  const float* b_cls     = (const float*)d_in[6];
  const float* w_bbox    = (const float*)d_in[7];
  const float* b_bbox    = (const float*)d_in[8];

  char* ws = (char*)d_ws;
  size_t off = 0;
  auto alloc = [&](size_t bytes) { size_t o = off; off += (bytes + 255) & ~(size_t)255; return o; };
  __bf16* Wb   = (__bf16*)(ws + alloc((size_t)512 * RPN_K3 * 2));
  __bf16* Wsm  = (__bf16*)(ws + alloc((size_t)64 * 512 * 2));
  float*  bsm  = (float*) (ws + alloc(64 * 4));
  __bf16* mid  = (__bf16*)(ws + alloc((size_t)512 * RPN_NPIX * 2));
  float*  sc   = (float*) (ws + alloc((size_t)64 * RPN_NPIX * 4));
  float*  boxA = (float*) (ws + alloc((size_t)4 * RPN_P * 4 * 4));
  float*  scoA = (float*) (ws + alloc((size_t)4 * RPN_P * 4));
  float*  topB = (float*) (ws + alloc((size_t)4 * RPN_PRE_N * 4 * 4));
  float*  topS = (float*) (ws + alloc((size_t)4 * RPN_PRE_N * 4));
  (void)ws_size; (void)in_sizes; (void)n_in; (void)out_size;

  k_pack_w3<<<(512 * RPN_K3 + 255) / 256, 256, 0, stream>>>(w_conv, Wb);
  k_pack_small<<<(64 * 512 + 255) / 256, 256, 0, stream>>>(w_cls, w_bbox, b_cls, b_bbox, Wsm, bsm);
  k_conv3<<<dim3(128, 4), 256, 0, stream>>>(base_feat, Wb, b_conv, mid);
  k_conv1<<<128, 256, 0, stream>>>(mid, Wsm, bsm, sc);
  k_props<<<(4 * RPN_P + 255) / 256, 256, 0, stream>>>(sc, im_shape, boxA, scoA);
  k_topk<<<4, 1024, 0, stream>>>(scoA, boxA, topS, topB);
  k_nms<<<4, 1024, 0, stream>>>(topS, topB, (float*)d_out);
}
